// Embedder_63299228009389
// MI455X (gfx1250) — compile-verified, matrix operand order checked by reading, not run on hardware
//
#include <hip/hip_runtime.h>
#include <hip/hip_bf16.h>

// Problem constants (from reference): B=64, T=256, DIM=512, H=512, L=3
#define B_   64
#define T_   256
#define DIM_ 512
#define H_   512
#define NWG_SCAN 96   // resident workgroups for the persistent scan kernels

typedef __attribute__((ext_vector_type(16))) __bf16 v16bf;
typedef __attribute__((ext_vector_type(8)))  __bf16 v8bf;
typedef __attribute__((ext_vector_type(8)))  float  v8f;

// D = A(16x32 bf16) x B(32x16 bf16) + C(16x16 f32)
#define WMMA_BF16(Af, Bf, Cf) \
  __builtin_amdgcn_wmma_f32_16x16x32_bf16(false, (Af), false, (Bf), (short)0, (Cf), false, false)

#if __has_builtin(__builtin_amdgcn_tensor_load_to_lds) && __has_builtin(__builtin_amdgcn_s_wait_tensorcnt)
#define HAVE_TDM 1
#else
#define HAVE_TDM 0
#endif

// Load one WMMA operand fragment from a row-major (stride in elements) bf16
// matrix tile. Matches the CDNA5 16-bit A/B 16x32 layout:
//   lanes 0-15 : row = lane,    K chunks [0..7] and [16..23]
//   lanes 16-31: row = lane-16, K chunks [8..15] and [24..31]
__device__ __forceinline__ v16bf load_frag(const __bf16* __restrict__ base,
                                           int stride, int lane) {
  const int r  = lane & 15;
  const int kb = (lane & 16) ? 8 : 0;
  const __bf16* p = base + r * stride + kb;
  v8bf lo = *(const v8bf*)(p);
  v8bf hi = *(const v8bf*)(p + 16);
  v16bf f;
#pragma unroll
  for (int i = 0; i < 8; ++i) { f[i] = lo[i]; f[i + 8] = hi[i]; }
  return f;
}

#if HAVE_TDM
typedef __attribute__((ext_vector_type(4))) unsigned tdm_u32x4;
typedef __attribute__((ext_vector_type(8))) int      tdm_i32x8;
typedef __attribute__((ext_vector_type(4))) int      tdm_i32x4;

// TDM: DMA a (rows_tile x 32 bf16) tile from a row-major bf16 tensor
// (row stride = K elements) into LDS with a 48-element (96B) row pitch:
// D# pad_interval = 16 DWORDs (code 3), pad_amount = 8 DWORDs (code 7).
// Toolchain here uses the 6-arg builtin: (u32x4, i32x8, i32x4, i32x4, i32x8, cpol).
__device__ __forceinline__ void tdm_load_tile_48p(const __bf16* g, unsigned lds_off,
                                                  int rows_total, int K, int rows_tile) {
  unsigned long long ga = (unsigned long long)(size_t)g;
  tdm_u32x4 g0;
  g0[0] = 1u;                                                 // count=1, user mode
  g0[1] = lds_off;                                            // LDS byte address
  g0[2] = (unsigned)ga;                                       // global addr [31:0]
  g0[3] = (unsigned)((ga >> 32) & 0x01ffffffu) | (2u << 30);  // addr[56:32] | type=2
  tdm_i32x8 g1;
  g1[0] = (1 << 16) | (1 << 20) | (3 << 22) | (7 << 25);      // 2B elems, pad 16dw->+8dw
  g1[1] = (int)(((unsigned)K & 0xffffu) << 16);               // tensor_dim0[15:0]
  g1[2] = (int)(((unsigned)K >> 16) | (((unsigned)rows_total & 0xffffu) << 16));
  g1[3] = (int)(((unsigned)rows_total >> 16) | (32u << 16));  // tile_dim0 = 32
  g1[4] = rows_tile;                                          // tile_dim1 (tile_dim2=0)
  g1[5] = K;                                                  // tensor_dim0_stride[31:0]
  g1[6] = 0;
  g1[7] = 0;
  tdm_i32x4 gz4 = {0, 0, 0, 0};
  tdm_i32x8 gz8 = {0, 0, 0, 0, 0, 0, 0, 0};
  __builtin_amdgcn_tensor_load_to_lds(g0, g1, gz4, gz4, gz8, 0);
}
#endif

// Device-wide barrier for the persistent scan kernels (NWG_SCAN workgroups).
__device__ __forceinline__ void grid_sync(unsigned* __restrict__ ctrl, unsigned phase) {
  __syncthreads();
  if (threadIdx.x == 0) {
    __threadfence();
    unsigned arrived = atomicAdd(ctrl, 1u) + 1u;
    if (arrived == (unsigned)NWG_SCAN) {
      atomicExch(ctrl, 0u);
      __threadfence();
      atomicAdd(ctrl + 1, 1u);
    } else {
      while (__hip_atomic_load(ctrl + 1, __ATOMIC_ACQUIRE, __HIP_MEMORY_SCOPE_AGENT) < phase)
        __builtin_amdgcn_s_sleep(1);
    }
    __threadfence();
  }
  __syncthreads();
}

__device__ __forceinline__ float sigf(float x) { return 1.f / (1.f + __expf(-x)); }

// ---------------------------------------------------------------- utilities
__global__ void k_zero_u32(unsigned* __restrict__ p, int n) {
  int i = blockIdx.x * blockDim.x + threadIdx.x;
  if (i < n) p[i] = 0u;
}

__global__ void k_cvt_bf16(const float* __restrict__ s, __bf16* __restrict__ d, int n) {
  for (int i = blockIdx.x * blockDim.x + threadIdx.x; i < n; i += gridDim.x * blockDim.x)
    d[i] = (__bf16)s[i];
}

// src: K x N (row-major f32)  ->  dst: N x K (row-major bf16)
__global__ void k_transpose_cvt(const float* __restrict__ s, __bf16* __restrict__ d,
                                int K, int N) {
  int total = K * N;
  for (int i = blockIdx.x * blockDim.x + threadIdx.x; i < total; i += gridDim.x * blockDim.x) {
    int k = i / N, n = i - k * N;
    d[(size_t)n * K + k] = (__bf16)s[i];
  }
}

// d(rows x 1024 bf16) = concat(a(rows x 512), b(rows x 512))
__global__ void k_concat_bf16(const float* __restrict__ a, const float* __restrict__ b,
                              __bf16* __restrict__ d, int rows) {
  int total = rows * 1024;
  for (int i = blockIdx.x * blockDim.x + threadIdx.x; i < total; i += gridDim.x * blockDim.x) {
    int r = i >> 10, c = i & 1023;
    float v = (c < 512) ? a[(size_t)r * 512 + c] : b[(size_t)r * 512 + (c - 512)];
    d[i] = (__bf16)v;
  }
}

// ----------------------------------------------------------- batched GEMM
// C(MxN f32) = A(MxK bf16, row-major) x Bt(NxK bf16, pre-transposed)^T + bias
// act: 0 = none, 1 = sigmoid.  M,N multiples of 64; K multiple of 32.
// WG = 256 thr = 8 waves; WG tile 64x64; wave (tm in 0..3, wn in 0..1) owns a
// 16x32 strip -> two 16x16 WMMA accumulators.
// Operand staging: TDM double-buffered LDS (tensor_load_to_lds + TENSORcnt)
// when available; per-thread global->LDS copies otherwise.
__global__ __launch_bounds__(256) void k_gemm_bf16(
    const __bf16* __restrict__ A, const __bf16* __restrict__ Bt,
    const float* __restrict__ bias, float* __restrict__ C,
    int M, int N, int K, int act)
{
  __shared__ __bf16 sA[2][64][48];   // 48-elem pitch: 16B aligned rows, conflict-free
  __shared__ __bf16 sB[2][64][48];
  const int tid  = threadIdx.x;
  const int lane = tid & 31;
  const int w    = tid >> 5;
  const int tm   = w & 3;
  const int wn   = w >> 2;
  const int m0   = blockIdx.y * 64;
  const int n0   = blockIdx.x * 64;
  v8f acc0 = {}; v8f acc1 = {};

#if HAVE_TDM
  const int nk = K >> 5;
  if (w == 0) {
    tdm_load_tile_48p(A  + (size_t)m0 * K, (unsigned)(size_t)&sA[0][0][0], M, K, 64);
    tdm_load_tile_48p(Bt + (size_t)n0 * K, (unsigned)(size_t)&sB[0][0][0], N, K, 64);
  }
  for (int kk = 0; kk < nk; ++kk) {
    if (w == 0) {
      if (kk + 1 < nk) {
        const int kn = (kk + 1) << 5;
        const int nb = (kk + 1) & 1;
        tdm_load_tile_48p(A  + (size_t)m0 * K + kn, (unsigned)(size_t)&sA[nb][0][0], M, K, 64);
        tdm_load_tile_48p(Bt + (size_t)n0 * K + kn, (unsigned)(size_t)&sB[nb][0][0], N, K, 64);
        __builtin_amdgcn_s_wait_tensorcnt(2);   // stage kk landed (in-order per wave)
      } else {
        __builtin_amdgcn_s_wait_tensorcnt(0);
      }
    }
    __syncthreads();
    const int cb = kk & 1;
    v16bf af = load_frag(&sA[cb][tm * 16][0], 48, lane);
    v16bf b0 = load_frag(&sB[cb][wn * 32][0], 48, lane);
    v16bf b1 = load_frag(&sB[cb][wn * 32 + 16][0], 48, lane);
    acc0 = WMMA_BF16(af, b0, acc0);
    acc1 = WMMA_BF16(af, b1, acc1);
    __syncthreads();                            // stage kk fully consumed
  }
#else
  const int lr = tid >> 2;        // 0..63 : tile row loaded by this thread
  const int lc = (tid & 3) * 8;   // 0,8,16,24
  for (int k0 = 0; k0 < K; k0 += 32) {
    v8bf av = *(const v8bf*)(A  + (size_t)(m0 + lr) * K + k0 + lc);
    v8bf bv = *(const v8bf*)(Bt + (size_t)(n0 + lr) * K + k0 + lc);
    __syncthreads();
    *(v8bf*)&sA[0][lr][lc] = av;
    *(v8bf*)&sB[0][lr][lc] = bv;
    __syncthreads();
    v16bf af = load_frag(&sA[0][tm * 16][0], 48, lane);
    v16bf b0 = load_frag(&sB[0][wn * 32][0], 48, lane);
    v16bf b1 = load_frag(&sB[0][wn * 32 + 16][0], 48, lane);
    acc0 = WMMA_BF16(af, b0, acc0);
    acc1 = WMMA_BF16(af, b1, acc1);
  }
#endif

  // C/D layout: lanes 0-15 -> N=lane, M=v; lanes 16-31 -> N=lane-16, M=8+v
  const int coll = lane & 15;
  const int rsub = (lane & 16) ? 8 : 0;
#pragma unroll
  for (int v = 0; v < 8; ++v) {
    const int grow = m0 + tm * 16 + rsub + v;
    int gcol = n0 + wn * 32 + coll;
    float val = acc0[v] + bias[gcol];
    if (act) val = sigf(val);
    C[(size_t)grow * N + gcol] = val;
    gcol = n0 + wn * 32 + 16 + coll;
    val = acc1[v] + bias[gcol];
    if (act) val = sigf(val);
    C[(size_t)grow * N + gcol] = val;
  }
}

// ------------------------------------------------------- persistent GRU scan
// xg: (B*T) x 1536 precomputed x@W + bi ; Ut: 1536 x 512 (bf16, transposed).
// Each wave owns one 16x16 tile of hg and keeps its 16 U-fragments (16x32 K
// slices) register-resident across all 256 timesteps: U is read exactly once.
__global__ __launch_bounds__(256) void k_gru_scan(
    const float* __restrict__ xg, const __bf16* __restrict__ Ut,
    const float* __restrict__ br, const int* __restrict__ lengths,
    float* __restrict__ out, float* __restrict__ h, __bf16* __restrict__ hbf,
    float* __restrict__ hg, unsigned* __restrict__ ctrl)
{
  const int NG   = 3 * H_;
  const int tid  = threadIdx.x;
  const int g    = blockIdx.x * 256 + tid;
  const int wid  = g >> 5;
  const int lane = tid & 31;
  unsigned phase = 0;

  const int  ntiles = 4 * (NG / 16);          // 384 tiles <= 768 waves
  const bool have   = (wid < ntiles);
  const int  tmm    = wid & 3;
  const int  tnn    = wid >> 2;
  v16bf bfrag[H_ / 32];
  if (have) {
    const __bf16* bbase = Ut + (size_t)tnn * 16 * H_;
#pragma unroll
    for (int kk = 0; kk < H_ / 32; ++kk)
      bfrag[kk] = load_frag(bbase + kk * 32, H_, lane);
  }

  for (int i = g; i < B_ * H_; i += NWG_SCAN * 256) { h[i] = 0.f; hbf[i] = (__bf16)0.f; }
  grid_sync(ctrl, ++phase);

  for (int t = 0; t < T_; ++t) {
    if (have) {                               // hg = h @ U (M=64, N=1536, K=512)
      const __bf16* abase = hbf + (size_t)tmm * 16 * H_;
      v8f acc = {};
#pragma unroll
      for (int kk = 0; kk < H_ / 32; ++kk)
        acc = WMMA_BF16(load_frag(abase + kk * 32, H_, lane), bfrag[kk], acc);
      const int col = tnn * 16 + (lane & 15);
      const int rb  = tmm * 16 + ((lane & 16) ? 8 : 0);
#pragma unroll
      for (int v = 0; v < 8; ++v) hg[(size_t)(rb + v) * NG + col] = acc[v];
    }
    grid_sync(ctrl, ++phase);

    // gates + state update (f32)
    for (int i = g; i < B_ * H_; i += NWG_SCAN * 256) {
      const int b = i >> 9;
      const int n = i & (H_ - 1);
      const float* xr  = xg + (size_t)(b * T_ + t) * NG;
      const float* hgp = hg + (size_t)b * NG;
      float z    = sigf(xr[n]          + hgp[n]          + br[n]);
      float r    = sigf(xr[n + H_]     + hgp[n + H_]     + br[n + H_]);
      float hh   = hgp[n + 2 * H_] + br[n + 2 * H_];
      float cand = tanhf(xr[n + 2 * H_] + r * hh);
      float hp   = h[i];
      float hn   = z * hp + (1.f - z) * cand;
      if (t >= lengths[b]) hn = hp;
      h[i]   = hn;
      hbf[i] = (__bf16)hn;
      out[(size_t)(b * T_ + t) * H_ + n] = hn;
    }
    grid_sync(ctrl, ++phase);
  }
}

// ------------------------------------------------------ persistent LSTM scan
// xg: (B*T) x 2048 precomputed x@W + b ; Ut: 2048 x 512 (bf16, transposed)
__global__ __launch_bounds__(256) void k_lstm_scan(
    const float* __restrict__ xg, const __bf16* __restrict__ Ut,
    const int* __restrict__ lengths, float* __restrict__ out,
    float* __restrict__ h, float* __restrict__ c, __bf16* __restrict__ hbf,
    float* __restrict__ hg, unsigned* __restrict__ ctrl)
{
  const int NG   = 4 * H_;
  const int tid  = threadIdx.x;
  const int g    = blockIdx.x * 256 + tid;
  const int wid  = g >> 5;
  const int lane = tid & 31;
  unsigned phase = 0;

  const int  ntiles = 4 * (NG / 16);          // 512 tiles <= 768 waves
  const bool have   = (wid < ntiles);
  const int  tmm    = wid & 3;
  const int  tnn    = wid >> 2;
  v16bf bfrag[H_ / 32];
  if (have) {
    const __bf16* bbase = Ut + (size_t)tnn * 16 * H_;
#pragma unroll
    for (int kk = 0; kk < H_ / 32; ++kk)
      bfrag[kk] = load_frag(bbase + kk * 32, H_, lane);
  }

  for (int i = g; i < B_ * H_; i += NWG_SCAN * 256) {
    h[i] = 0.f; c[i] = 0.f; hbf[i] = (__bf16)0.f;
  }
  grid_sync(ctrl, ++phase);

  for (int t = 0; t < T_; ++t) {
    if (have) {
      const __bf16* abase = hbf + (size_t)tmm * 16 * H_;
      v8f acc = {};
#pragma unroll
      for (int kk = 0; kk < H_ / 32; ++kk)
        acc = WMMA_BF16(load_frag(abase + kk * 32, H_, lane), bfrag[kk], acc);
      const int col = tnn * 16 + (lane & 15);
      const int rb  = tmm * 16 + ((lane & 16) ? 8 : 0);
#pragma unroll
      for (int v = 0; v < 8; ++v) hg[(size_t)(rb + v) * NG + col] = acc[v];
    }
    grid_sync(ctrl, ++phase);

    for (int i = g; i < B_ * H_; i += NWG_SCAN * 256) {
      const int b = i >> 9;
      const int n = i & (H_ - 1);
      const float* xr  = xg + (size_t)(b * T_ + t) * NG;
      const float* hgp = hg + (size_t)b * NG;
      float iv = sigf(xr[n]          + hgp[n]);
      float fv = sigf(xr[n + H_]     + hgp[n + H_]);
      float gc =      xr[n + 2 * H_] + hgp[n + 2 * H_];
      float ov = sigf(xr[n + 3 * H_] + hgp[n + 3 * H_]);
      float cp = c[i];
      float hp = h[i];
      float cn = fv * cp + iv * tanhf(gc);
      float hn = ov * tanhf(cn);
      if (t >= lengths[b]) { hn = hp; cn = cp; }
      h[i]   = hn;
      c[i]   = cn;
      hbf[i] = (__bf16)hn;
      out[(size_t)(b * T_ + t) * H_ + n] = hn;
    }
    grid_sync(ctrl, ++phase);
  }
}

// --------------------------------------------------------------------- host
extern "C" void kernel_launch(void* const* d_in, const int* in_sizes, int n_in,
                              void* d_out, int out_size, void* d_ws, size_t ws_size,
                              hipStream_t stream) {
  (void)in_sizes; (void)n_in; (void)out_size; (void)ws_size;
  const float* x       = (const float*)d_in[0];
  const int*   lengths = (const int*)d_in[1];
  const float* gru_W   = (const float*)d_in[2];
  const float* gru_U   = (const float*)d_in[3];
  const float* gru_b   = (const float*)d_in[4];
  const float* lstm_W  = (const float*)d_in[5];
  const float* lstm_U  = (const float*)d_in[6];
  const float* lstm_b  = (const float*)d_in[7];
  const float* dense_W = (const float*)d_in[8];
  const float* dense_b = (const float*)d_in[9];
  float* out = (float*)d_out;

  const int ROWS = B_ * T_;                 // 16384
  char* p = (char*)d_ws;
  auto take = [&](size_t bytes) -> char* {
    char* r = p; p += (bytes + 255) & ~(size_t)255; return r;
  };
  unsigned* ctrl    = (unsigned*)take(4096);                 // 6 barrier blocks, 256B apart
  __bf16* gruWt     = (__bf16*)take((size_t)3 * 1536 * 512 * 2);
  __bf16* gruUt     = (__bf16*)take((size_t)3 * 1536 * 512 * 2);
  __bf16* lstmWt    = (__bf16*)take((size_t)3 * 2048 * 512 * 2);
  __bf16* lstmUt    = (__bf16*)take((size_t)3 * 2048 * 512 * 2);
  __bf16* denseWt   = (__bf16*)take((size_t)512 * 1024 * 2);
  __bf16* xbf       = (__bf16*)take((size_t)ROWS * 1024 * 2);
  float*  xg        = (float*)take((size_t)ROWS * 2048 * 4);
  float*  hg        = (float*)take((size_t)B_ * 2048 * 4);
  float*  hst       = (float*)take((size_t)B_ * H_ * 4);
  float*  cst       = (float*)take((size_t)B_ * H_ * 4);
  __bf16* hbf       = (__bf16*)take((size_t)B_ * H_ * 2);
  float*  bufA      = (float*)take((size_t)ROWS * H_ * 4);
  float*  bufB      = (float*)take((size_t)ROWS * H_ * 4);
  float*  bufC      = (float*)take((size_t)ROWS * H_ * 4);

  k_zero_u32<<<4, 256, 0, stream>>>(ctrl, 1024);

  // Weight prep: transpose + bf16 (N x K layout).
  for (int l = 0; l < 3; ++l) {
    k_transpose_cvt<<<2048, 256, 0, stream>>>(gru_W  + (size_t)l * 512 * 1536,
                                              gruWt  + (size_t)l * 1536 * 512, 512, 1536);
    k_transpose_cvt<<<2048, 256, 0, stream>>>(gru_U  + (size_t)l * 512 * 1536,
                                              gruUt  + (size_t)l * 1536 * 512, 512, 1536);
    k_transpose_cvt<<<2048, 256, 0, stream>>>(lstm_W + (size_t)l * 512 * 2048,
                                              lstmWt + (size_t)l * 2048 * 512, 512, 2048);
    k_transpose_cvt<<<2048, 256, 0, stream>>>(lstm_U + (size_t)l * 512 * 2048,
                                              lstmUt + (size_t)l * 2048 * 512, 512, 2048);
  }
  k_transpose_cvt<<<2048, 256, 0, stream>>>(dense_W, denseWt, 1024, 512);

  // ---- GRU stack: x -> bufA -> bufB -> bufA
  {
    const float* cur = x;
    float* outs[3] = {bufA, bufB, bufA};
    for (int l = 0; l < 3; ++l) {
      k_cvt_bf16<<<4096, 256, 0, stream>>>(cur, xbf, ROWS * 512);
      k_gemm_bf16<<<dim3(1536 / 64, ROWS / 64), 256, 0, stream>>>(
          xbf, gruWt + (size_t)l * 1536 * 512,
          gru_b + (size_t)l * 2 * 1536,          // bi
          xg, ROWS, 1536, 512, 0);
      k_gru_scan<<<NWG_SCAN, 256, 0, stream>>>(
          xg, gruUt + (size_t)l * 1536 * 512,
          gru_b + (size_t)l * 2 * 1536 + 1536,   // br
          lengths, outs[l], hst, hbf, hg, ctrl + l * 64);
      cur = outs[l];
    }
  }

  // ---- LSTM stack: x -> bufB -> bufC -> bufB  (bufA keeps GRU result)
  {
    const float* cur = x;
    float* outs[3] = {bufB, bufC, bufB};
    for (int l = 0; l < 3; ++l) {
      k_cvt_bf16<<<4096, 256, 0, stream>>>(cur, xbf, ROWS * 512);
      k_gemm_bf16<<<dim3(2048 / 64, ROWS / 64), 256, 0, stream>>>(
          xbf, lstmWt + (size_t)l * 2048 * 512,
          lstm_b + (size_t)l * 2048,             // full bias once
          xg, ROWS, 2048, 512, 0);
      k_lstm_scan<<<NWG_SCAN, 256, 0, stream>>>(
          xg, lstmUt + (size_t)l * 2048 * 512,
          lengths, outs[l], hst, cst, hbf, hg, ctrl + (3 + l) * 64);
      cur = outs[l];
    }
  }

  // ---- final dense: sigmoid(concat(o1,o2) @ dense_W + dense_b)
  k_concat_bf16<<<4096, 256, 0, stream>>>(bufA, bufB, xbf, ROWS);
  k_gemm_bf16<<<dim3(512 / 64, ROWS / 64), 256, 0, stream>>>(
      xbf, denseWt, dense_b, out, ROWS, 512, 1024, 1);
}
